// GaussianImage_Cholesky_39779987095872
// MI455X (gfx1250) — compile-verified, matrix operand order checked by reading, not run on hardware
//
#include <hip/hip_runtime.h>
#include <hip/hip_bf16.h>

typedef __attribute__((ext_vector_type(16))) _Float16 v16h;
typedef __attribute__((ext_vector_type(8)))  float    v8f;

#define GN    512   // number of gaussians
#define IMG   512   // image H == W
#define TILES 32    // 512 / 16
#define MAXG  512   // max compacted gaussians per tile (padded count <= 512)

// ---------------------------------------------------------------------------
// Kernel 1: per-gaussian preprocessing (conic, center, tile bounds)
// ---------------------------------------------------------------------------
__global__ void gi_preprocess(const float* __restrict__ xyz,
                              const float* __restrict__ chol,
                              const float* __restrict__ opac,
                              float* __restrict__ P,     // [6][GN]: ca cb cc cx cy op
                              int*   __restrict__ T) {   // [4][GN]: tminx tmaxx tminy tmaxy
  int g = blockIdx.x * blockDim.x + threadIdx.x;
  if (g >= GN) return;

  float mx = tanhf(xyz[g * 2 + 0]);
  float my = tanhf(xyz[g * 2 + 1]);
  float l0 = chol[g * 3 + 0] + 0.5f;
  float l1 = chol[g * 3 + 1];
  float l2 = chol[g * 3 + 2] + 0.5f;

  float cxx = l0 * l0;
  float cxy = l0 * l1;
  float cyy = l1 * l1 + l2 * l2;
  float det = cxx * cyy - cxy * cxy;
  float inv = 1.0f / det;
  float ca  =  cyy * inv;
  float cb  = -cxy * inv;
  float cc  =  cxx * inv;

  float cx = 0.5f * (float)IMG * (mx + 1.0f);
  float cy = 0.5f * (float)IMG * (my + 1.0f);

  float b  = 0.5f * (cxx + cyy);
  float v1 = b + sqrtf(fmaxf(b * b - det, 0.1f));
  float radius = ceilf(3.0f * sqrtf(v1));

  int tminx = min(max((int)((cx - radius) / 16.0f), 0), TILES);
  int tmaxx = min(max((int)((cx + radius) / 16.0f) + 1, 0), TILES);
  int tminy = min(max((int)((cy - radius) / 16.0f), 0), TILES);
  int tmaxy = min(max((int)((cy + radius) / 16.0f) + 1, 0), TILES);

  P[0 * GN + g] = ca;
  P[1 * GN + g] = cb;
  P[2 * GN + g] = cc;
  P[3 * GN + g] = cx;
  P[4 * GN + g] = cy;
  P[5 * GN + g] = opac[g];
  T[0 * GN + g] = tminx;
  T[1 * GN + g] = tmaxx;
  T[2 * GN + g] = tminy;
  T[3 * GN + g] = tmaxy;
}

// ---------------------------------------------------------------------------
// Kernel 2: one 256-thread block per 16x16 tile.
//   - wave 0 compacts active gaussians (order-preserving -> deterministic)
//   - each wave owns 2 tile rows; alpha evaluated in VALU, channel einsum
//     done with v_wmma_f32_16x16x32_f16 (A = features 16x32 f16 padded 3->16,
//     B = alpha 32x16 f16, D = f32 16x16 accumulator)
// ---------------------------------------------------------------------------
__global__ void __launch_bounds__(256)
gi_render(const float* __restrict__ P,
          const int*   __restrict__ T,
          const float* __restrict__ feat,   // [GN][3]
          float*       __restrict__ out) {  // [3][IMG][IMG]
  __shared__ float    s_ca[MAXG], s_cb[MAXG], s_cc[MAXG];
  __shared__ float    s_cx[MAXG], s_cy[MAXG], s_op[MAXG];
  __shared__ _Float16 s_fh[MAXG * 4];       // 4 slots per gaussian, slot 3 == 0
  __shared__ int      s_cnt;

  const int tid  = threadIdx.x;
  const int lane = tid & 31;
  const int wid  = tid >> 5;
  const int tx   = blockIdx.x;
  const int ty   = blockIdx.y;

  // ---- deterministic per-tile compaction by wave 0 ----
  if (wid == 0) {
    const int* tminx = T;
    const int* tmaxx = T + GN;
    const int* tminy = T + 2 * GN;
    const int* tmaxy = T + 3 * GN;
    int cnt = 0;
    for (int base = 0; base < GN; base += 32) {
      int  g   = base + lane;
      bool act = (tx >= tminx[g]) && (tx < tmaxx[g]) &&
                 (ty >= tminy[g]) && (ty < tmaxy[g]);
      unsigned long long m = __ballot(act);
      if (act) {
        int pos = cnt + (int)__popcll(m & ((1ull << lane) - 1ull));
        s_ca[pos] = P[0 * GN + g];
        s_cb[pos] = P[1 * GN + g];
        s_cc[pos] = P[2 * GN + g];
        s_cx[pos] = P[3 * GN + g];
        s_cy[pos] = P[4 * GN + g];
        s_op[pos] = P[5 * GN + g];
        s_fh[pos * 4 + 0] = (_Float16)feat[g * 3 + 0];
        s_fh[pos * 4 + 1] = (_Float16)feat[g * 3 + 1];
        s_fh[pos * 4 + 2] = (_Float16)feat[g * 3 + 2];
        s_fh[pos * 4 + 3] = (_Float16)0.0f;
      }
      cnt += (int)__popcll(m);
    }
    // pad to a multiple of 32 with zero-opacity entries (alpha == 0 exactly)
    int cntPad = (cnt + 31) & ~31;
    for (int p = cnt + lane; p < cntPad; p += 32) {
      s_ca[p] = 0.0f; s_cb[p] = 0.0f; s_cc[p] = 0.0f;
      s_cx[p] = 0.0f; s_cy[p] = 0.0f; s_op[p] = 0.0f;
      s_fh[p * 4 + 0] = (_Float16)0.0f;
      s_fh[p * 4 + 1] = (_Float16)0.0f;
      s_fh[p * 4 + 2] = (_Float16)0.0f;
      s_fh[p * 4 + 3] = (_Float16)0.0f;
    }
    if (lane == 0) s_cnt = cntPad;
  }
  __syncthreads();
  const int total = s_cnt;

  // ---- per-lane geometry ----
  const int   n    = lane & 15;      // pixel column within tile / A-matrix row (channel)
  const int   half = lane >> 4;      // lane half selects K sub-range
  const int   cc4  = (n < 3) ? n : 3;  // feature slot (slot 3 is zero => padded channels)
  const float px   = (float)(tx * 16 + n);
  const float py0  = (float)(ty * 16 + 2 * wid);
  const float py1  = py0 + 1.0f;

  v8f acc0 = {};   // tile row 2*wid
  v8f acc1 = {};   // tile row 2*wid + 1

  for (int base = 0; base < total; base += 32) {
    v16h a, b0, b1;
#pragma unroll
    for (int v = 0; v < 8; ++v) {
      // 16-bit A/B VGPR striping: VGPR v holds K = k0, k0+1 for this lane
      const int k0 = 2 * (v & 3) + 8 * half + 16 * (v >> 2);
#pragma unroll
      for (int t = 0; t < 2; ++t) {
        const int k = base + k0 + t;
        // A: features (channel = lane%16, zero-padded beyond channel 2)
        a[2 * v + t] = s_fh[k * 4 + cc4];
        // B: alpha for this lane's pixel column, two adjacent tile rows
        const float ga  = s_ca[k];
        const float gb  = s_cb[k];
        const float gc  = s_cc[k];
        const float op  = s_op[k];
        const float dx  = s_cx[k] - px;
        const float dy0 = s_cy[k] - py0;
        const float dy1 = s_cy[k] - py1;
        const float hx  = 0.5f * ga * dx * dx;
        const float s0  = hx + 0.5f * gc * dy0 * dy0 + gb * dx * dy0;
        const float s1  = hx + 0.5f * gc * dy1 * dy1 + gb * dx * dy1;
        b0[2 * v + t] = (_Float16)(__expf(-s0) * op);
        b1[2 * v + t] = (_Float16)(__expf(-s1) * op);
      }
    }
    // D(16 channels x 16 pixels) += A(16x32 f16) x B(32x16 f16), f32 accum
    acc0 = __builtin_amdgcn_wmma_f32_16x16x32_f16(
        false, a, false, b0, (short)0, acc0, false, false);
    acc1 = __builtin_amdgcn_wmma_f32_16x16x32_f16(
        false, a, false, b1, (short)0, acc1, false, false);
  }

  // ---- store: D VGPR r, lanes 0..15 hold [channel=r][pixel n]; rows 3..15 are 0
  if (half == 0) {
    const int x  = tx * 16 + n;
    const int y0 = ty * 16 + 2 * wid;
    const int y1 = y0 + 1;
    const int HW = IMG * IMG;
#pragma unroll
    for (int ch = 0; ch < 3; ++ch) {
      float v0 = fminf(fmaxf(acc0[ch], 0.0f), 1.0f);
      float v1 = fminf(fmaxf(acc1[ch], 0.0f), 1.0f);
      out[ch * HW + y0 * IMG + x] = v0;
      out[ch * HW + y1 * IMG + x] = v1;
    }
  }
}

// ---------------------------------------------------------------------------
extern "C" void kernel_launch(void* const* d_in, const int* in_sizes, int n_in,
                              void* d_out, int out_size, void* d_ws, size_t ws_size,
                              hipStream_t stream) {
  const float* xyz  = (const float*)d_in[0];  // [GN][2]
  const float* chol = (const float*)d_in[1];  // [GN][3]
  const float* feat = (const float*)d_in[2];  // [GN][3]
  const float* opac = (const float*)d_in[3];  // [GN][1]

  float* P = (float*)d_ws;              // 6*GN floats
  int*   T = (int*)(P + 6 * GN);        // 4*GN ints

  gi_preprocess<<<dim3((GN + 255) / 256), dim3(256), 0, stream>>>(
      xyz, chol, opac, P, T);

  dim3 grid(TILES, TILES);
  gi_render<<<grid, dim3(256), 0, stream>>>(P, T, feat, (float*)d_out);
}
